// DeepSeekSparseAttention_53386443489531
// MI455X (gfx1250) — matrix operand order repaired and checked by reading, not compiled, not run on hardware
//
#include <hip/hip_runtime.h>

typedef _Float16 half_t;
typedef __attribute__((ext_vector_type(16))) _Float16 v16h;
typedef __attribute__((ext_vector_type(8)))  float    v8f;

#define S_LEN   2048
#define HEADS   12
#define DH      64
#define DMODEL  768
#define TOPK_C  256
#define W2      256          // LOCAL_WINDOW/2
#define SCALE_C 0.125f       // 1/sqrt(64)

__device__ __forceinline__ v8f wmma_f16(const v16h& a, const v16h& b, const v8f& c) {
    // D = A(16x32 f16) * B(32x16 f16) + C(16x16 f32)
    return __builtin_amdgcn_wmma_f32_16x16x32_f16(
        /*neg_a=*/false, a, /*neg_b=*/false, b,
        /*c_mod=*/(short)0, c, /*reuse_a=*/false, /*reuse_b=*/false);
}

__device__ __forceinline__ float rowmax16(float v) {
    v = fmaxf(v, __shfl_xor(v, 1, 32));
    v = fmaxf(v, __shfl_xor(v, 2, 32));
    v = fmaxf(v, __shfl_xor(v, 4, 32));
    v = fmaxf(v, __shfl_xor(v, 8, 32));
    return v;
}
__device__ __forceinline__ float rowsum16(float v) {
    v += __shfl_xor(v, 1, 32);
    v += __shfl_xor(v, 2, 32);
    v += __shfl_xor(v, 4, 32);
    v += __shfl_xor(v, 8, 32);
    return v;
}

// ---------------------------------------------------------------------------
// CDNA5 async memory -> LDS staging (tracked with ASYNCcnt).
// One 32-key K chunk = 4KB contiguous; 8 issues x (32 lanes x 16B).
// Low 32 bits of a generic pointer into __shared__ are the LDS byte offset
// ---------------------------------------------------------------------------
__device__ __forceinline__ void issue_async_chunk(half_t* dst_lds, const half_t* src) {
    const int lane = threadIdx.x & 31;
    const unsigned base = (unsigned)(size_t)(void*)dst_lds;
#pragma unroll
    for (int it = 0; it < 8; ++it) {
        const unsigned loff = base + it * 512 + lane * 16;
        const half_t* g = src + it * 256 + lane * 8;
        asm volatile("global_load_async_to_lds_b128 %0, %1, off"
                     :: "v"(loff), "v"(g)
                     : "memory");
    }
}
__device__ __forceinline__ void wait_async_le8() {
    asm volatile("s_wait_asynccnt 8" ::: "memory");
}
__device__ __forceinline__ void wait_async_0() {
    asm volatile("s_wait_asynccnt 0" ::: "memory");
}

// ---------------------------------------------------------------------------
// Projection GEMM: C[M=2048, N=768] = X(f32) @ W(f32)^T + bias, result f16.
// Destination indexed as head-major with configurable strides so the same
// kernel writes Q/K row-major [H][S][64] or V transposed [H][64][S].
// ---------------------------------------------------------------------------
__global__ void __launch_bounds__(128)
gemm_qkv_kernel(const float* __restrict__ X, const float* __restrict__ W,
                const float* __restrict__ bias, half_t* __restrict__ dst,
                int s_stride, int d_stride)
{
    const int wave = threadIdx.x >> 5;
    const int lane = threadIdx.x & 31;
    const int m0 = blockIdx.x * 16;
    const int n0 = blockIdx.y * 64 + wave * 16;
    const int n  = lane & 15;
    const int hh = lane >> 4;
    const int m  = lane & 15;

    v8f acc = {};
    const float* arow = X + (size_t)(m0 + m) * DMODEL;
    const float* brow = W + (size_t)(n0 + n) * DMODEL;

    for (int kc = 0; kc < DMODEL; kc += 32) {
        v16h a, b;
#pragma unroll
        for (int e = 0; e < 16; ++e) {
            const int g = e >> 3, o = e & 7;
            a[e] = (half_t)arow[kc + hh * 16 + e];   // A[m][k]
            b[e] = (half_t)brow[kc + hh * 16 + e];           // B[k][n] = W[n][k]
        }
        acc = wmma_f16(a, b, acc);
    }

    const float bn  = bias[n0 + n];
    const int head  = (n0 + n) >> 6;
    const int d     = (n0 + n) & 63;
    half_t* dh = dst + (size_t)head * (S_LEN * DH);
#pragma unroll
    for (int r = 0; r < 8; ++r) {
        const int srow = m0 + r + 8 * hh;
        dh[(size_t)srow * s_stride + (size_t)d * d_stride] = (half_t)(acc[r] + bn);
    }
}

// ---------------------------------------------------------------------------
// token_scores[h][s] = dot(x[s], Ws[h]) + bs[h]
// ---------------------------------------------------------------------------
__global__ void __launch_bounds__(256)
token_score_kernel(const float* __restrict__ X, const float* __restrict__ Ws,
                   const float* __restrict__ bs, float* __restrict__ ts)
{
    const int t = blockIdx.x * blockDim.x + threadIdx.x;
    if (t >= HEADS * S_LEN) return;
    const int h = t / S_LEN, s = t % S_LEN;
    const float* xr = X  + (size_t)s * DMODEL;
    const float* wr = Ws + (size_t)h * DMODEL;
    float acc = bs[h];
    for (int k = 0; k < DMODEL; ++k) acc = fmaf(xr[k], wr[k], acc);
    ts[h * S_LEN + s] = acc;
}

// ---------------------------------------------------------------------------
// Top-k selection via rank counting (matches lax.top_k stable tie-breaking):
// column j selected iff #{j' : v[j']>v[j]} + #{j'<j : v[j']==v[j]} < 256.
// ---------------------------------------------------------------------------
__global__ void __launch_bounds__(256)
topk_flags_kernel(const float* __restrict__ ts, unsigned int* __restrict__ flags)
{
    __shared__ float sv[S_LEN];
    const int h = blockIdx.x;
    for (int i = threadIdx.x; i < S_LEN; i += 256) sv[i] = ts[h * S_LEN + i];
    __syncthreads();
    for (int j = threadIdx.x; j < S_LEN; j += 256) {
        const float vj = sv[j];
        int rank = 0;
        for (int p = 0; p < S_LEN; ++p) {
            const float vp = sv[p];
            rank += (vp > vj) || ((vp == vj) && (p < j));
        }
        flags[h * S_LEN + j] = (rank < TOPK_C) ? 1u : 0u;
    }
}

// ---------------------------------------------------------------------------
// Prefix-scan the flags, compact selected indices in ascending order, and
// gather dense K_sel [H][256][64] (row-major) and V_sel^T [H][64][256].
// ---------------------------------------------------------------------------
__global__ void __launch_bounds__(256)
topk_gather_kernel(const unsigned int* __restrict__ flags,
                   const half_t* __restrict__ kb, const half_t* __restrict__ vT,
                   half_t* __restrict__ Ksel, half_t* __restrict__ VselT)
{
    __shared__ int buf0[S_LEN];
    __shared__ int buf1[S_LEN];
    __shared__ int selidx[TOPK_C];
    const int h = blockIdx.x;

    for (int i = threadIdx.x; i < S_LEN; i += 256) buf0[i] = (int)flags[h * S_LEN + i];
    __syncthreads();

    int* src = buf0;
    int* dst = buf1;
    for (int off = 1; off < S_LEN; off <<= 1) {
        for (int i = threadIdx.x; i < S_LEN; i += 256)
            dst[i] = src[i] + ((i >= off) ? src[i - off] : 0);
        __syncthreads();
        int* tmp = src; src = dst; dst = tmp;
    }
    // src = inclusive prefix sum of flags
    for (int i = threadIdx.x; i < S_LEN; i += 256) {
        const int incl = src[i];
        const int fl   = incl - (i ? src[i - 1] : 0);
        if (fl && incl <= TOPK_C) selidx[incl - 1] = i;
    }
    __syncthreads();

    const int p = threadIdx.x;
    if (p < TOPK_C) {
        const int j = selidx[p];
        // K row copy (64 halves, contiguous)
        const unsigned int* srow = (const unsigned int*)(kb + ((size_t)h * S_LEN + j) * DH);
        unsigned int* drow = (unsigned int*)(Ksel + ((size_t)h * TOPK_C + p) * DH);
#pragma unroll
        for (int d = 0; d < DH / 2; ++d) drow[d] = srow[d];
        // V^T column gather (strided)
        const half_t* vcol = vT    + (size_t)h * DH * S_LEN  + j;
        half_t*       vdst = VselT + (size_t)h * DH * TOPK_C + p;
#pragma unroll 8
        for (int d = 0; d < DH; ++d) vdst[(size_t)d * TOPK_C] = vcol[(size_t)d * S_LEN];
    }
}

// ---------------------------------------------------------------------------
// Flash-attention branch state: per-wave 16-row query tile, 64-wide output.
// C-layout: lane holds rows r+8*hh (r=0..7), column = lane&15 of each n-tile.
// ---------------------------------------------------------------------------
struct BState {
    float m[8];
    float l[8];
    v8f   o[4];
};

__device__ __forceinline__ void bstate_init(BState& st) {
#pragma unroll
    for (int r = 0; r < 8; ++r) { st.m[r] = -1e30f; st.l[r] = 0.f; }
    v8f z = {};
#pragma unroll
    for (int t = 0; t < 4; ++t) st.o[t] = z;
}

// ksrc points at the 32-row chunk base (LDS staging buffer or global memory).
// vkey0 indexes V^T; (q0g, key0g) drive the mask.
// mode: 0 = no mask (top-k), 1 = local band |i-j|<=256, 2 = global j<16
__device__ __forceinline__ void attn_chunk(
    BState& st, const v16h& qa0, const v16h& qa1,
    const half_t* __restrict__ ksrc,   // chunk: [32][64] row-major
    const half_t* __restrict__ vsrcT,  // [64][vstride] (transposed V)
    int vstride, int vkey0, int q0g, int key0g, int mode,
    half_t* ldsP)                      // 16*32 f16 staging for this wave
{
    const int lane = threadIdx.x & 31;
    const int n  = lane & 15;
    const int hh = lane >> 4;

    // ---- scores: S(16x32) = Q(16x64) * K_chunk(32x64)^T, two 16-col subtiles
    v8f s0 = {}, s1 = {};
    {
        const half_t* kr0 = ksrc + (size_t)n        * DH + hh * 16;
        const half_t* kr1 = ksrc + (size_t)(16 + n) * DH + hh * 16;
        v16h b;
        b = *(const v16h*)(kr0);      s0 = wmma_f16(qa0, b, s0);
        b = *(const v16h*)(kr0 + 32); s0 = wmma_f16(qa1, b, s0);
        b = *(const v16h*)(kr1);      s1 = wmma_f16(qa0, b, s1);
        b = *(const v16h*)(kr1 + 32); s1 = wmma_f16(qa1, b, s1);
    }

    // ---- scale + mask + online softmax statistics
    float sc[8];
#pragma unroll
    for (int r = 0; r < 8; ++r) {
        const int i  = q0g + r + 8 * hh;
        const int j0 = key0g + n;
        const int j1 = key0g + 16 + n;
        float v0 = s0[r] * SCALE_C;
        float v1 = s1[r] * SCALE_C;
        bool ok0 = true, ok1 = true;
        if (mode == 1) {
            const int d0 = i - j0, d1 = i - j1;
            ok0 = (d0 <= W2) && (d0 >= -W2);
            ok1 = (d1 <= W2) && (d1 >= -W2);
        } else if (mode == 2) {
            ok0 = (j0 < 16);
            ok1 = (j1 < 16);
        }
        v0 = ok0 ? v0 : -1e30f;
        v1 = ok1 ? v1 : -1e30f;

        const float mc = rowmax16(fmaxf(v0, v1));
        const float mn = fmaxf(st.m[r], mc);
        sc[r] = (st.m[r] > -1e29f) ? __expf(st.m[r] - mn) : 0.f;
        const float p0 = (v0 > -1e29f) ? __expf(v0 - mn) : 0.f;
        const float p1 = (v1 > -1e29f) ? __expf(v1 - mn) : 0.f;
        s0[r] = p0;
        s1[r] = p1;
        st.l[r] = st.l[r] * sc[r] + rowsum16(p0 + p1);
        st.m[r] = mn;
    }
#pragma unroll
    for (int t = 0; t < 4; ++t)
#pragma unroll
        for (int r = 0; r < 8; ++r)
            st.o[t][r] *= sc[r];

    // ---- P (f32 C-layout) -> f16 A-layout via LDS cross-lane permute
#pragma unroll
    for (int r = 0; r < 8; ++r) {
        ldsP[(r + 8 * hh) * 32 + n]      = (half_t)s0[r];
        ldsP[(r + 8 * hh) * 32 + 16 + n] = (half_t)s1[r];
    }
    __builtin_amdgcn_wave_barrier();
    asm volatile("" ::: "memory");
    v16h pa;
    const int mm = lane & 15;
#pragma unroll
    for (int e = 0; e < 16; ++e) {
        const int g = e >> 3, o = e & 7;
        pa[e] = ldsP[mm * 32 + hh * 16 + e];
    }
    __builtin_amdgcn_wave_barrier();

    // ---- O += P(16x32) * V_chunk(32x64); B-tiles contiguous from V^T
#pragma unroll
    for (int t = 0; t < 4; ++t) {
        const v16h vb = *(const v16h*)(vsrcT + (size_t)(t * 16 + n) * vstride + vkey0 + hh * 16);
        st.o[t] = wmma_f16(pa, vb, st.o[t]);
    }
}

__device__ __forceinline__ void bstate_finalize(const BState& st, float comb[4][8]) {
    float inv[8];
#pragma unroll
    for (int r = 0; r < 8; ++r) inv[r] = (1.0f / 3.0f) / (st.l[r] + 1e-20f);
#pragma unroll
    for (int t = 0; t < 4; ++t)
#pragma unroll
        for (int r = 0; r < 8; ++r) comb[t][r] += st.o[t][r] * inv[r];
}

// ---------------------------------------------------------------------------
// Fused 3-branch attention: one wave per 16-row query tile; 4 waves/block.
// K chunks double-buffered in LDS via global_load_async_to_lds_b128.
// ---------------------------------------------------------------------------
__global__ void __launch_bounds__(128)
attn_kernel(const half_t* __restrict__ qb, const half_t* __restrict__ kb,
            const half_t* __restrict__ vT, const half_t* __restrict__ Ksel,
            const half_t* __restrict__ VselT, half_t* __restrict__ attn_out)
{
    __shared__ half_t ldsK[4][2][32 * DH];   // per-wave double buffer, 2x4KB
    __shared__ half_t ldsP[4][16 * 32];      // per-wave P permute stage, 1KB
    const int wave = threadIdx.x >> 5;
    const int lane = threadIdx.x & 31;
    const int h  = blockIdx.y;
    const int q0 = (blockIdx.x * 4 + wave) * 16;
    const int n  = lane & 15;
    const int hh = lane >> 4;
    const int m  = lane & 15;
    half_t* myP = ldsP[wave];
    half_t* kst = &ldsK[wave][0][0];

    // Load Q tile into WMMA A-layout (two K=32 slabs)
    const half_t* qrow = qb + ((size_t)h * S_LEN + q0) * DH;
    v16h qa0, qa1;
#pragma unroll
    for (int e = 0; e < 16; ++e) {
        const int g = e >> 3, o = e & 7;
        qa0[e] = qrow[(size_t)m * DH + hh * 16 + e];
        qa1[e] = qrow[(size_t)m * DH + 32 + hh * 16 + e];
    }

    float comb[4][8];
#pragma unroll
    for (int t = 0; t < 4; ++t)
#pragma unroll
        for (int r = 0; r < 8; ++r) comb[t][r] = 0.f;

    const half_t* kh = kb + (size_t)h * S_LEN * DH;
    const half_t* vh = vT + (size_t)h * DH * S_LEN;

    BState st;

    // ---- Branch 1: local band, async double-buffered K stream
    bstate_init(st);
    {
        int jlo = q0 - W2; if (jlo < 0) jlo = 0; jlo &= ~31;
        int jhi = q0 + 15 + W2 + 1; if (jhi > S_LEN) jhi = S_LEN;
        issue_async_chunk(kst, kh + (size_t)jlo * DH);
        for (int key0 = jlo; key0 < jhi; key0 += 32) {
            const int c = (key0 - jlo) >> 5;
            half_t* cur = kst + (c & 1) * (32 * DH);
            if (key0 + 32 < jhi) {
                issue_async_chunk(kst + ((c + 1) & 1) * (32 * DH),
                                  kh + (size_t)(key0 + 32) * DH);
                __builtin_prefetch(vh + key0 + 32, 0, 0);
                wait_async_le8();
            } else {
                wait_async_0();
            }
            attn_chunk(st, qa0, qa1, cur, vh, S_LEN, key0, q0, key0, 1, myP);
        }
    }
    bstate_finalize(st, comb);

    // ---- Branch 2: top-k (dense over compacted 256 keys), async staged
    bstate_init(st);
    {
        const half_t* ks = Ksel  + (size_t)h * TOPK_C * DH;
        const half_t* vs = VselT + (size_t)h * DH * TOPK_C;
        issue_async_chunk(kst, ks);
        for (int key0 = 0; key0 < TOPK_C; key0 += 32) {
            const int c = key0 >> 5;
            half_t* cur = kst + (c & 1) * (32 * DH);
            if (key0 + 32 < TOPK_C) {
                issue_async_chunk(kst + ((c + 1) & 1) * (32 * DH),
                                  ks + (size_t)(key0 + 32) * DH);
                wait_async_le8();
            } else {
                wait_async_0();
            }
            attn_chunk(st, qa0, qa1, cur, vs, TOPK_C, key0, q0, key0, 0, myP);
        }
    }
    bstate_finalize(st, comb);

    // ---- Branch 3: global (first 16 keys), single direct chunk
    bstate_init(st);
    attn_chunk(st, qa0, qa1, kh, vh, S_LEN, 0, q0, 0, 2, myP);
    bstate_finalize(st, comb);

    // ---- Store pre-projection context [S][768] as f16
#pragma unroll
    for (int t = 0; t < 4; ++t)
#pragma unroll
        for (int r = 0; r < 8; ++r)
            attn_out[(size_t)(q0 + r + 8 * hh) * DMODEL + h * DH + t * 16 + n] =
                (half_t)comb[t][r];
}

// ---------------------------------------------------------------------------
// Output projection: out[S][768] = attn(f16) @ Wo^T + bo, f32 result.
// ---------------------------------------------------------------------------
__global__ void __launch_bounds__(128)
gemm_oproj_kernel(const half_t* __restrict__ A, const float* __restrict__ W,
                  const float* __restrict__ bias, float* __restrict__ out)
{
    const int wave = threadIdx.x >> 5;
    const int lane = threadIdx.x & 31;
    const int m0 = blockIdx.x * 16;
    const int n0 = blockIdx.y * 64 + wave * 16;
    const int n  = lane & 15;
    const int hh = lane >> 4;
    const int m  = lane & 15;

    v8f acc = {};
    const half_t* arow = A + (size_t)(m0 + m) * DMODEL;
    const float*  brow = W + (size_t)(n0 + n) * DMODEL;

    for (int kc = 0; kc < DMODEL; kc += 32) {
        v16h a, b;
#pragma unroll
        for (int e = 0; e < 16; ++e) {
            const int g = e >> 3, o = e & 7;
            a[e] = arow[kc + hh * 16 + e];
            b[e] = (half_t)brow[kc + hh * 16 + e];
        }
        acc = wmma_f16(a, b, acc);
    }

    const float bn = bias[n0 + n];
#pragma unroll
    for (int r = 0; r < 8; ++r)
        out[(size_t)(m0 + r + 8 * hh) * DMODEL + n0 + n] = acc[r] + bn;
}

// ---------------------------------------------------------------------------
extern "C" void kernel_launch(void* const* d_in, const int* in_sizes, int n_in,
                              void* d_out, int out_size, void* d_ws, size_t ws_size,
                              hipStream_t stream)
{
    const float* x  = (const float*)d_in[0];
    const float* Wq = (const float*)d_in[1];
    const float* bq = (const float*)d_in[2];
    const float* Wk = (const float*)d_in[3];
    const float* bk = (const float*)d_in[4];
    const float* Wv = (const float*)d_in[5];
    const float* bv = (const float*)d_in[6];
    const float* Wo = (const float*)d_in[7];
    const float* bo = (const float*)d_in[8];
    const float* Ws = (const float*)d_in[9];
    const float* bs = (const float*)d_in[10];
    float* out = (float*)d_out;

    char* ws = (char*)d_ws;
    size_t off = 0;
    auto walloc = [&](size_t bytes) -> char* {
        char* p = ws + off;
        off = (off + bytes + 255) & ~(size_t)255;
        return p;
    };
    half_t* qb    = (half_t*)walloc((size_t)HEADS * S_LEN * DH * 2);   // [H][S][64]
    half_t* kbuf  = (half_t*)walloc((size_t)HEADS * S_LEN * DH * 2);   // [H][S][64]
    half_t* vTbuf = (half_t*)walloc((size_t)HEADS * DH * S_LEN * 2);   // [H][64][S]
    float*  ts    = (float*) walloc((size_t)HEADS * S_LEN * 4);        // [H][S]
    unsigned int* flags = (unsigned int*)walloc((size_t)HEADS * S_LEN * 4);
    half_t* Ksel  = (half_t*)walloc((size_t)HEADS * TOPK_C * DH * 2);  // [H][256][64]
    half_t* VselT = (half_t*)walloc((size_t)HEADS * DH * TOPK_C * 2);  // [H][64][256]
    half_t* attnb = (half_t*)walloc((size_t)S_LEN * DMODEL * 2);       // [S][768]

    dim3 gproj(S_LEN / 16, DMODEL / 64);   // (128, 12)
    dim3 bproj(128);

    // Q, K row-major; V transposed for contiguous PV B-tiles
    gemm_qkv_kernel<<<gproj, bproj, 0, stream>>>(x, Wq, bq, qb,    DH, 1);
    gemm_qkv_kernel<<<gproj, bproj, 0, stream>>>(x, Wk, bk, kbuf,  DH, 1);
    gemm_qkv_kernel<<<gproj, bproj, 0, stream>>>(x, Wv, bv, vTbuf, 1,  S_LEN);

    token_score_kernel<<<(HEADS * S_LEN + 255) / 256, 256, 0, stream>>>(x, Ws, bs, ts);
    topk_flags_kernel<<<HEADS, 256, 0, stream>>>(ts, flags);
    topk_gather_kernel<<<HEADS, 256, 0, stream>>>(flags, kbuf, vTbuf, Ksel, VselT);

    attn_kernel<<<dim3(S_LEN / 64, HEADS), 128, 0, stream>>>(qb, kbuf, vTbuf,
                                                             Ksel, VselT, attnb);

    gemm_oproj_kernel<<<gproj, bproj, 0, stream>>>(attnb, Wo, bo, out);
}
